// MusicRNN_70007966925171
// MI455X (gfx1250) — matrix operand order repaired and checked by reading, not compiled, not run on hardware
//
#include <hip/hip_runtime.h>

// ---------------------------------------------------------------------------
// Fused 2-layer LSTM + FC head for MI455X (gfx1250, wave32, WMMA bf16)
//   grid  = B/16 workgroups, each owns 16 batch rows for all T steps
//   block = 256 threads = 8 waves; each wave computes two 16-wide gate tiles
// ---------------------------------------------------------------------------

typedef __attribute__((ext_vector_type(16))) __bf16 bf16x16;
typedef __attribute__((ext_vector_type(8)))  __bf16 bf16x8;
typedef __attribute__((ext_vector_type(8)))  float  f32x8;

constexpr int Bsz = 2048;
constexpr int T   = 256;
constexpr int I   = 88;
constexpr int IP  = 96;          // K padded to multiple of 32 for WMMA
constexpr int H   = 64;
constexpr int G4  = 4 * H;       // 256 gate outputs (i,f,g,o)
constexpr int O   = 13;
constexpr int MT  = 16;          // batch rows per workgroup
constexpr int NTHREADS = 256;

// ---- LDS layout (all offsets/sizes 16B aligned) ----
constexpr size_t OFF_WIH0  = 0;
constexpr size_t OFF_WHH0  = OFF_WIH0  + (size_t)G4 * IP * 2;   // 49152
constexpr size_t OFF_WIH1  = OFF_WHH0  + (size_t)G4 * H  * 2;   // +32768
constexpr size_t OFF_WHH1  = OFF_WIH1  + (size_t)G4 * H  * 2;
constexpr size_t OFF_XA    = OFF_WHH1  + (size_t)G4 * H  * 2;
constexpr size_t OFF_H1B   = OFF_XA    + (size_t)MT * IP * 2;
constexpr size_t OFF_H2B   = OFF_H1B   + (size_t)MT * H  * 2;
constexpr size_t OFF_C1    = OFF_H2B   + (size_t)MT * H  * 2;
constexpr size_t OFF_C2    = OFF_C1    + (size_t)MT * H  * 4;
constexpr size_t OFF_H2F   = OFF_C2    + (size_t)MT * H  * 4;
constexpr size_t OFF_GATES = OFF_H2F   + (size_t)MT * H  * 4;
constexpr size_t OFF_B0    = OFF_GATES + (size_t)MT * G4 * 4;
constexpr size_t OFF_B1    = OFF_B0    + (size_t)G4 * 4;
constexpr size_t OFF_FCW   = OFF_B1    + (size_t)G4 * 4;
constexpr size_t OFF_FCB   = OFF_FCW   + (size_t)O * H * 4;
constexpr size_t LDS_BYTES = OFF_FCB   + 64;                    // ~184.6 KB

// ---- fast transcendentals: v_exp_f32 + v_rcp_f32, no IEEE division ----
__device__ __forceinline__ float fast_sigmoid(float v) {
    return __builtin_amdgcn_rcpf(1.f + __expf(-v));
}
__device__ __forceinline__ float fast_tanh(float v) {
    // tanh(v) = 2*sigmoid(2v) - 1
    return 2.f * fast_sigmoid(v + v) - 1.f;
}

// A fragment (16x32 bf16): lane L -> row m = L&15, hi = L>>4.
// v16 element j maps to K = (j<8 ? j : j+8) + 8*hi  => two contiguous 8-elem
// (16-byte) chunks at k0+8*hi and k0+16+8*hi.
__device__ __forceinline__ bf16x16 frag_a(const __bf16* base, int stride,
                                          int k0, int lane) {
    const int m  = lane & 15;
    const int hi = lane >> 4;
    const __bf16* p = base + m * stride + k0 + 8 * hi;
    bf16x8 lo = *(const bf16x8*)(p);
    bf16x8 up = *(const bf16x8*)(p + 16);
    return __builtin_shufflevector(lo, up, 0, 1, 2, 3, 4, 5, 6, 7,
                                           8, 9, 10, 11, 12, 13, 14, 15);
}

// B fragment (32x16 bf16): lane L -> col n = L&15, hi = L>>4,
// element j -> K = (j<8 ? j : j+8) + 8*hi. Weights are stored row-major [gate][K] which is
// exactly B^T, so lane n reads two 8-elem K chunks from its weight row.
__device__ __forceinline__ bf16x16 frag_b(const __bf16* wrows, int stride,
                                          int k0, int lane) {
    const int n  = lane & 15;
    const int hi = lane >> 4;
    const __bf16* p = wrows + n * stride + k0 + 8 * hi;
    bf16x8 lo = *(const bf16x8*)(p);
    bf16x8 up = *(const bf16x8*)(p + 16);
    return __builtin_shufflevector(lo, up, 0, 1, 2, 3, 4, 5, 6, 7,
                                           8, 9, 10, 11, 12, 13, 14, 15);
}

// One LSTM gate stage: gates[16][256] = act( A_x @ Wx^T + A_h @ Wh^T + bias )
// Two independent accumulators (x-path, h-path) halve the WMMA D->C chain.
template <int AK>
__device__ __forceinline__ void gate_stage(const __bf16* ax, const __bf16* wx,
                                           const __bf16* ah, const __bf16* wh,
                                           const float* bias, float* gates,
                                           int wave, int lane) {
#pragma unroll
    for (int s = 0; s < 2; ++s) {
        const int nt = wave * 2 + s;
        const int g0 = nt * 16;
        f32x8 accx = {};
        f32x8 acch = {};
#pragma unroll
        for (int kk = 0; kk < AK; kk += 32) {
            bf16x16 a = frag_a(ax, AK, kk, lane);
            bf16x16 b = frag_b(wx + (size_t)g0 * AK, AK, kk, lane);
            accx = __builtin_amdgcn_wmma_f32_16x16x32_bf16(
                false, a, false, b, (short)0, accx, false, false);
        }
#pragma unroll
        for (int kk = 0; kk < H; kk += 32) {
            bf16x16 a = frag_a(ah, H, kk, lane);
            bf16x16 b = frag_b(wh + (size_t)g0 * H, H, kk, lane);
            acch = __builtin_amdgcn_wmma_f32_16x16x32_bf16(
                false, a, false, b, (short)0, acch, false, false);
        }
        // C/D layout: lane -> n = lane&15; VGPR r -> m = r + 8*(lane>>4)
        const int n   = lane & 15;
        const int hi  = lane >> 4;
        const int g   = g0 + n;
        const bool isg = ((g0 >> 6) == 2);   // gate quadrant 2 = cell gate
        const float bb = bias[g];
#pragma unroll
        for (int r = 0; r < 8; ++r) {
            float v = accx[r] + acch[r] + bb;
            // branch-free: tanh(v) = 2*sigmoid(2v)-1
            float u = isg ? (v + v) : v;
            float sg = fast_sigmoid(u);
            float res = isg ? (2.f * sg - 1.f) : sg;
            gates[(r + 8 * hi) * G4 + g] = res;
        }
    }
}

// c' = f*c + i*g ; h' = o*tanh(c') ; h stored as bf16 (GEMM) and opt. f32
__device__ __forceinline__ void state_update(float* cst, __bf16* hb, float* hf,
                                             const float* gates, int tid) {
#pragma unroll
    for (int idx0 = 0; idx0 < MT * H; idx0 += NTHREADS) {
        const int idx = idx0 + tid;
        const int m = idx >> 6;
        const int h = idx & 63;
        const float* gr = gates + (size_t)m * G4;
        const float c  = gr[H + h] * cst[idx] + gr[h] * gr[2 * H + h];
        const float hn = gr[3 * H + h] * fast_tanh(c);
        cst[idx] = c;
        hb[idx]  = (__bf16)hn;
        if (hf) hf[idx] = hn;
    }
}

__global__ void __launch_bounds__(NTHREADS, 1)
lstm2_fused_kernel(const float* __restrict__ x,
                   const float* __restrict__ w_ih0, const float* __restrict__ w_hh0,
                   const float* __restrict__ b_ih0, const float* __restrict__ b_hh0,
                   const float* __restrict__ w_ih1, const float* __restrict__ w_hh1,
                   const float* __restrict__ b_ih1, const float* __restrict__ b_hh1,
                   const float* __restrict__ fc_w, const float* __restrict__ fc_b,
                   float* __restrict__ out) {
    extern __shared__ char smem[];
    __bf16* wih0  = (__bf16*)(smem + OFF_WIH0);
    __bf16* whh0  = (__bf16*)(smem + OFF_WHH0);
    __bf16* wih1  = (__bf16*)(smem + OFF_WIH1);
    __bf16* whh1  = (__bf16*)(smem + OFF_WHH1);
    __bf16* xa    = (__bf16*)(smem + OFF_XA);
    __bf16* h1b   = (__bf16*)(smem + OFF_H1B);
    __bf16* h2b   = (__bf16*)(smem + OFF_H2B);
    float*  c1    = (float*)(smem + OFF_C1);
    float*  c2    = (float*)(smem + OFF_C2);
    float*  h2f   = (float*)(smem + OFF_H2F);
    float*  gates = (float*)(smem + OFF_GATES);
    float*  b0s   = (float*)(smem + OFF_B0);
    float*  b1s   = (float*)(smem + OFF_B1);
    float*  fcw   = (float*)(smem + OFF_FCW);
    float*  fcb   = (float*)(smem + OFF_FCB);

    const int tid   = threadIdx.x;
    const int lane  = tid & 31;
    const int wave  = tid >> 5;
    const int bbase = blockIdx.x * MT;

    // ---- one-time: convert weights fp32 -> bf16 into LDS ----
    for (int idx = tid; idx < G4 * IP; idx += NTHREADS) {
        const int g = idx / IP, k = idx % IP;
        wih0[idx] = (k < I) ? (__bf16)w_ih0[(size_t)g * I + k] : (__bf16)0.f;
    }
    for (int idx = tid; idx < G4 * H; idx += NTHREADS) {
        whh0[idx] = (__bf16)w_hh0[idx];
        wih1[idx] = (__bf16)w_ih1[idx];
        whh1[idx] = (__bf16)w_hh1[idx];
    }
    for (int idx = tid; idx < G4; idx += NTHREADS) {
        b0s[idx] = b_ih0[idx] + b_hh0[idx];
        b1s[idx] = b_ih1[idx] + b_hh1[idx];
    }
    for (int idx = tid; idx < O * H; idx += NTHREADS) fcw[idx] = fc_w[idx];
    if (tid < O) fcb[tid] = fc_b[tid];
    for (int idx = tid; idx < MT * H; idx += NTHREADS) {
        c1[idx] = 0.f; c2[idx] = 0.f; h2f[idx] = 0.f;
        h1b[idx] = (__bf16)0.f; h2b[idx] = (__bf16)0.f;
    }

    // t-invariant staging coordinates: 16*96/256 = 6 elements per thread
    int sm[6], sk[6];
#pragma unroll
    for (int j = 0; j < 6; ++j) {
        const int idx = j * NTHREADS + tid;
        sm[j] = idx / IP;
        sk[j] = idx % IP;
    }
    __syncthreads();

    // ---- recurrence over time, both layers fused ----
    for (int t = 0; t < T; ++t) {
        // stage x tile [16][96] (bf16, zero-padded K)
#pragma unroll
        for (int j = 0; j < 6; ++j) {
            const int m = sm[j], k = sk[j];
            const float* gp = x + ((size_t)(bbase + m) * T + t) * I + k;
            xa[j * NTHREADS + tid] = (k < I) ? (__bf16)(*gp) : (__bf16)0.f;
            // prefetch next timestep's tile while this step computes
            if (t + 1 < T && k < I)
                __builtin_prefetch(gp + I, 0, 0);
        }
        __syncthreads();

        // layer 1: gates = act(x@Wih0^T + h1@Whh0^T + b0)
        gate_stage<IP>(xa, wih0, h1b, whh0, b0s, gates, wave, lane);
        __syncthreads();
        state_update(c1, h1b, nullptr, gates, tid);
        __syncthreads();

        // layer 2: gates = act(h1@Wih1^T + h2@Whh1^T + b1)
        gate_stage<H>(h1b, wih1, h2b, whh1, b1s, gates, wave, lane);
        __syncthreads();
        state_update(c2, h2b, h2f, gates, tid);
        __syncthreads();
    }

    // ---- FC head on final h2 ----
    if (tid < MT * O) {
        const int m = tid / O, o = tid % O;
        float s = fcb[o];
#pragma unroll 8
        for (int k = 0; k < H; ++k) s += fcw[o * H + k] * h2f[m * H + k];
        out[(size_t)(bbase + m) * O + o] = s;
    }
}

extern "C" void kernel_launch(void* const* d_in, const int* in_sizes, int n_in,
                              void* d_out, int out_size, void* d_ws, size_t ws_size,
                              hipStream_t stream) {
    (void)in_sizes; (void)n_in; (void)out_size; (void)d_ws; (void)ws_size;
    const float* x     = (const float*)d_in[0];
    const float* w_ih0 = (const float*)d_in[1];
    const float* w_hh0 = (const float*)d_in[2];
    const float* b_ih0 = (const float*)d_in[3];
    const float* b_hh0 = (const float*)d_in[4];
    const float* w_ih1 = (const float*)d_in[5];
    const float* w_hh1 = (const float*)d_in[6];
    const float* b_ih1 = (const float*)d_in[7];
    const float* b_hh1 = (const float*)d_in[8];
    const float* fc_w  = (const float*)d_in[9];
    const float* fc_b  = (const float*)d_in[10];
    float* out = (float*)d_out;

    static const int _attr_once = []() {
        return (int)hipFuncSetAttribute(
            (const void*)lstm2_fused_kernel,
            hipFuncAttributeMaxDynamicSharedMemorySize, (int)LDS_BYTES);
    }();
    (void)_attr_once;

    dim3 grid(Bsz / MT);
    dim3 block(NTHREADS);
    lstm2_fused_kernel<<<grid, block, LDS_BYTES, stream>>>(
        x, w_ih0, w_hh0, b_ih0, b_hh0,
        w_ih1, w_hh1, b_ih1, b_hh1, fc_w, fc_b, out);
}
